// GATRegressor_18107582119952
// MI455X (gfx1250) — matrix operand order repaired and checked by reading, not compiled, not run on hardware
//
#include <hip/hip_runtime.h>
#include <hip/hip_bf16.h>
#include <math.h>

// ---------------------------------------------------------------------------
// GAT regressor for MI455X (gfx1250, wave32, WMMA).
//   N=50000 nodes, E=1.6M edges (+N self loops), F=HID=128, G=512 graphs.
// Dense x@W uses v_wmma_f32_16x16x32_f16 with LDS staged in *fragment order*
// so operand loads are contiguous ds_load_b128, not scalar u16 gathers.
// Scatter parts use f32 global atomics (HBM-bound majority of the work).
// ---------------------------------------------------------------------------

#define NF 128  // feature width (F_IN == HID == 128)

typedef __attribute__((ext_vector_type(16))) _Float16 v16h;
typedef __attribute__((ext_vector_type(8)))  float    v8f;

__device__ __forceinline__ float lrelu(float x) { return x > 0.f ? x : 0.2f * x; }

__device__ __forceinline__ unsigned pack2h(float a, float b) {
    union { _Float16 h[2]; unsigned u; } p;
    p.h[0] = (_Float16)a;
    p.h[1] = (_Float16)b;
    return p.u;
}

// Monotonic float atomic-max via ordered-int trick (lowers to
// global_atomic_max_i32 / global_atomic_min_u32).
__device__ __forceinline__ void atomicMaxF(float* addr, float v) {
    if (v >= 0.f) atomicMax((int*)addr, __float_as_int(v));
    else          atomicMin((unsigned int*)addr, __float_as_uint(v));
}

// ---------------------------------------------------------------------------
// fill kernel (re-initializes workspace every launch; capture-safe)
// ---------------------------------------------------------------------------
__global__ void fill_f32(float* __restrict__ p, float v, long n) {
    long stride = (long)gridDim.x * blockDim.x;
    for (long i = (long)blockIdx.x * blockDim.x + threadIdx.x; i < n; i += stride)
        p[i] = v;
}

// ---------------------------------------------------------------------------
// WMMA GEMM:  Y[nrows x 128] = act(X[nrows x 128]) @ W[128 x 128]
//   - relu_in != 0 applies ReLU to X while staging (layer-1 fuses relu(x)).
//   - block = 128 threads (4 waves), 64 rows per block.
//   - LDS staged in WMMA *fragment order* (wave32; contiguous k, not the ISA table order):
//       A(16x32 f16): lane l: M=l&15; lanes 0-15 hold K=elem,
//                     16-31 K=elem+16.
//       B(32x16 f16): lane l: N=l&15; lanes 0-15 hold K=elem, 16-31 K=elem+16.
//     so each operand is one contiguous 32B v16h per lane -> ds_load_b128 x2.
//   - per wave: 16 rows x 128 cols, K=128 in 4 steps of 32 -> 32 WMMAs.
// ---------------------------------------------------------------------------
__global__ __launch_bounds__(128) void gemm_relu_wmma(
    const float* __restrict__ X, const float* __restrict__ W,
    float* __restrict__ Y, int nrows, int relu_in) {
    __shared__ v16h lxf[4 * 4 * 32];   // [wave][ks][lane] : 16KB
    __shared__ v16h lwf[4 * 8 * 32];   // [ks][nt][lane]   : 32KB
    unsigned* lxu = (unsigned*)lxf;
    unsigned* lwu = (unsigned*)lwf;

    const int tid   = threadIdx.x;
    const int rbase = blockIdx.x * 64;

    // ---- stage W (128x128) into B-fragment order, packed 2xf16 stores ----
    // pair index p = kpair*128 + n ; k = 2*kpair
    for (int i = 0; i < 64; ++i) {
        int p  = tid + i * 128;
        int kp = p >> 7, n = p & (NF - 1);
        int k  = kp * 2;
        float w0 = W[(long)k * NF + n];
        float w1 = W[(long)(k + 1) * NF + n];
        int ks = k >> 5, kr = k & 31;
        int nt = n >> 4, nlo = n & 15;
        int lane = nlo + ((kr >> 4) << 4);   // +16 for K>=16
        int v    = (kr & 15) >> 1;           // packed-pair index (kr even)
        lwu[(((ks * 8 + nt) * 32) + lane) * 8 + v] = pack2h(w0, w1);
    }

    // ---- stage X tile (64x128, optional ReLU) into A-fragment order ----
    // pair index p = r*64 + kpair ; k = 2*kpair
    for (int i = 0; i < 32; ++i) {
        int p  = tid + i * 128;
        int r  = p >> 6, kp = p & 63;
        int k  = kp * 2;
        int gr = rbase + r;
        float x0 = 0.f, x1 = 0.f;
        if (gr < nrows) {
            x0 = X[(long)gr * NF + k];
            x1 = X[(long)gr * NF + k + 1];
        }
        if (relu_in) {
            x0 = x0 > 0.f ? x0 : 0.f;
            x1 = x1 > 0.f ? x1 : 0.f;
        }
        int wv = r >> 4, m = r & 15;
        int ks = k >> 5, kr = k & 31;
        int laneHi = kr >> 4;
        int lane   = m + laneHi * 16;
        int elem   = kr & 15;   // even
        lxu[(((wv * 4 + ks) * 32) + lane) * 8 + (elem >> 1)] = pack2h(x0, x1);
    }
    __syncthreads();

    const int lane   = tid & 31;
    const int wave   = tid >> 5;
    const int laneLo = lane & 15;
    const int laneHi = lane >> 4;

    v8f acc[8] = {};

#pragma unroll
    for (int ks = 0; ks < 4; ++ks) {
        v16h a = lxf[(wave * 4 + ks) * 32 + lane];
#pragma unroll
        for (int nt = 0; nt < 8; ++nt) {
            v16h b = lwf[(ks * 8 + nt) * 32 + lane];
            acc[nt] = __builtin_amdgcn_wmma_f32_16x16x32_f16(
                false, a, false, b, (short)0, acc[nt], false, false);
        }
    }

    // C/D layout: vreg r: M = r + laneHi*8, N = laneLo
#pragma unroll
    for (int nt = 0; nt < 8; ++nt) {
#pragma unroll
        for (int r = 0; r < 8; ++r) {
            int m    = r + laneHi * 8;
            int grow = rbase + wave * 16 + m;
            if (grow < nrows)
                Y[(long)grow * NF + nt * 16 + laneLo] = acc[nt][r];
        }
    }
}

// ---------------------------------------------------------------------------
// Per-node attention scores: asrc[i] = h[i].a_src, adst[i] = h[i].a_dst
// One wave per node; float4 per lane (32*4 = 128) + shuffle reduce.
// ---------------------------------------------------------------------------
__global__ __launch_bounds__(256) void node_scores_k(
    const float* __restrict__ H, const float* __restrict__ av_s,
    const float* __restrict__ av_d, float* __restrict__ outs,
    float* __restrict__ outd, int n) {
    int wave = (int)((blockIdx.x * (long)blockDim.x + threadIdx.x) >> 5);
    int lane = threadIdx.x & 31;
    int nw   = (int)(((long)gridDim.x * blockDim.x) >> 5);
    for (int node = wave; node < n; node += nw) {
        float4 hv = ((const float4*)(H + (long)node * NF))[lane];
        float4 sa = ((const float4*)av_s)[lane];
        float4 da = ((const float4*)av_d)[lane];
        float s = hv.x * sa.x + hv.y * sa.y + hv.z * sa.z + hv.w * sa.w;
        float d = hv.x * da.x + hv.y * da.y + hv.z * da.z + hv.w * da.w;
        for (int off = 16; off > 0; off >>= 1) {
            s += __shfl_down(s, off, 32);
            d += __shfl_down(d, off, 32);
        }
        if (lane == 0) { outs[node] = s; outd[node] = d; }
    }
}

// ---------------------------------------------------------------------------
// Edge softmax pass 1: per-dst running max of leaky_relu logits.
// Edges e < E come from edge_index; e >= E are the N self loops.
// ---------------------------------------------------------------------------
__global__ void edge_max_k(const int* __restrict__ ei,
                           const float* __restrict__ asrc,
                           const float* __restrict__ adst,
                           float* __restrict__ maxd, int E_, int n) {
    long tot    = (long)E_ + n;
    long stride = (long)gridDim.x * blockDim.x;
    for (long e = (long)blockIdx.x * blockDim.x + threadIdx.x; e < tot; e += stride) {
        int s, d;
        if (e < (long)E_) { s = ei[e]; d = ei[(long)E_ + e]; }
        else              { s = d = (int)(e - E_); }
        atomicMaxF(&maxd[d], lrelu(asrc[s] + adst[d]));
    }
}

// Edge softmax pass 2: ex[e] = exp(logit - max[dst]); denom[dst] += ex[e]
__global__ void edge_exp_k(const int* __restrict__ ei,
                           const float* __restrict__ asrc,
                           const float* __restrict__ adst,
                           const float* __restrict__ maxd,
                           float* __restrict__ exbuf,
                           float* __restrict__ denom, int E_, int n) {
    long tot    = (long)E_ + n;
    long stride = (long)gridDim.x * blockDim.x;
    for (long e = (long)blockIdx.x * blockDim.x + threadIdx.x; e < tot; e += stride) {
        int s, d;
        if (e < (long)E_) { s = ei[e]; d = ei[(long)E_ + e]; }
        else              { s = d = (int)(e - E_); }
        float ex = __expf(lrelu(asrc[s] + adst[d]) - maxd[d]);
        exbuf[e] = ex;
        atomicAdd(&denom[d], ex);
    }
}

// Edge aggregation: agg[dst] += h[src] * (ex[e] / denom[dst]).
// One wave per edge; float4 gather per lane, 4 f32 atomic adds per lane.
__global__ __launch_bounds__(256) void edge_agg_k(
    const int* __restrict__ ei, const float* __restrict__ H,
    const float* __restrict__ exbuf, const float* __restrict__ denom,
    float* __restrict__ agg, int E_, int n) {
    long tot  = (long)E_ + n;
    long wid  = (blockIdx.x * (long)blockDim.x + threadIdx.x) >> 5;
    int  lane = threadIdx.x & 31;
    long nw   = ((long)gridDim.x * blockDim.x) >> 5;
    for (long e = wid; e < tot; e += nw) {
        int s, d;
        if (e < (long)E_) { s = ei[e]; d = ei[(long)E_ + e]; }
        else              { s = d = (int)(e - E_); }
        float alpha = exbuf[e] / denom[d];
        float4 hv   = ((const float4*)(H + (long)s * NF))[lane];
        float* ap   = agg + (long)d * NF + lane * 4;
        atomicAdd(ap + 0, hv.x * alpha);
        atomicAdd(ap + 1, hv.y * alpha);
        atomicAdd(ap + 2, hv.z * alpha);
        atomicAdd(ap + 3, hv.w * alpha);
    }
}

// h = relu(agg + bias), in place
__global__ void bias_relu_k(float* __restrict__ h, const float* __restrict__ b,
                            long n128) {
    long stride = (long)gridDim.x * blockDim.x;
    for (long i = (long)blockIdx.x * blockDim.x + threadIdx.x; i < n128; i += stride) {
        float v = h[i] + b[i & (NF - 1)];
        h[i] = v > 0.f ? v : 0.f;
    }
}

// per-graph node counts
__global__ void count_k(const int* __restrict__ batch, float* __restrict__ cnt,
                        int n) {
    long stride = (long)gridDim.x * blockDim.x;
    for (long i = (long)blockIdx.x * blockDim.x + threadIdx.x; i < n; i += stride)
        atomicAdd(&cnt[batch[i]], 1.0f);
}

// pooling scatter: maxp[g][f] = max(...), sump[g][f] += h
__global__ void pool_k(const float* __restrict__ h, const int* __restrict__ batch,
                       float* __restrict__ maxp, float* __restrict__ sump, int n) {
    long tot    = (long)n * NF;
    long stride = (long)gridDim.x * blockDim.x;
    for (long i = (long)blockIdx.x * blockDim.x + threadIdx.x; i < tot; i += stride) {
        int node = (int)(i >> 7), f = (int)(i & (NF - 1));
        int g = batch[node];
        float v = h[i];
        atomicMaxF(&maxp[(long)g * NF + f], v);
        atomicAdd(&sump[(long)g * NF + f], v);
    }
}

// final: out[g] = [maxp(g), meanp(g)] . fc_w + fc_b   (one block per graph)
__global__ __launch_bounds__(256) void final_k(
    const float* __restrict__ maxp, const float* __restrict__ sump,
    const float* __restrict__ cnt, const float* __restrict__ fcw,
    const float* __restrict__ fcb, float* __restrict__ out) {
    __shared__ float red[256];
    int g = blockIdx.x;
    int t = threadIdx.x;
    float c = cnt[g];
    float val;
    if (t < NF) {
        float m = maxp[(long)g * NF + t];
        if (!(c > 0.f)) m = 0.f;
        val = m * fcw[t];
    } else {
        int f = t - NF;
        float mean = sump[(long)g * NF + f] / fmaxf(c, 1.f);
        val = mean * fcw[t];
    }
    red[t] = val;
    __syncthreads();
    for (int off = 128; off > 0; off >>= 1) {
        if (t < off) red[t] += red[t + off];
        __syncthreads();
    }
    if (t == 0) out[g] = red[0] + fcb[0];
}

// ---------------------------------------------------------------------------
// launch
// ---------------------------------------------------------------------------
extern "C" void kernel_launch(void* const* d_in, const int* in_sizes, int n_in,
                              void* d_out, int out_size, void* d_ws, size_t ws_size,
                              hipStream_t stream) {
    const float* x      = (const float*)d_in[0];
    const int*   ei     = (const int*)d_in[1];
    const int*   batch  = (const int*)d_in[2];
    const float* W1     = (const float*)d_in[3];
    const float* a_src1 = (const float*)d_in[4];
    const float* a_dst1 = (const float*)d_in[5];
    const float* b1     = (const float*)d_in[6];
    const float* W2     = (const float*)d_in[7];
    const float* a_src2 = (const float*)d_in[8];
    const float* a_dst2 = (const float*)d_in[9];
    const float* b2     = (const float*)d_in[10];
    const float* fc_w   = (const float*)d_in[11];
    const float* fc_b   = (const float*)d_in[12];
    float*       out    = (float*)d_out;

    const int N_ = in_sizes[0] / NF;   // 50000
    const int E_ = in_sizes[1] / 2;    // 1600000
    const int G_ = out_size;           // 512
    const long EE = (long)E_ + N_;

    // workspace carve-up (floats)
    float* bufA  = (float*)d_ws;                 // N x 128  (h_lin both layers)
    float* bufB  = bufA  + (size_t)N_ * NF;      // N x 128  (agg -> h post act)
    float* asrc  = bufB  + (size_t)N_ * NF;      // N
    float* adst  = asrc  + N_;                   // N
    float* maxd  = adst  + N_;                   // N
    float* denom = maxd  + N_;                   // N
    float* exbuf = denom + N_;                   // E + N
    float* maxp  = exbuf + EE;                   // G x 128
    float* sump  = maxp  + (size_t)G_ * NF;      // G x 128
    float* cnt   = sump  + (size_t)G_ * NF;      // G

    const dim3 blk(256);
    const float NEG_INF = -__builtin_inff();
    const int gemmBlocks  = (N_ + 63) / 64;
    const int scoreBlocks = (N_ * 32 + 255) / 256;
    const int edgeBlocks  = (int)((EE + 255) / 256);
    const int eaggBlocks  = (int)((EE * 32 + 255) / 256);

    // ---- layer 1 -----------------------------------------------------------
    fill_f32<<<2048, blk, 0, stream>>>(bufB, 0.f, (long)N_ * NF);
    fill_f32<<<512,  blk, 0, stream>>>(maxd, NEG_INF, N_);
    fill_f32<<<512,  blk, 0, stream>>>(denom, 0.f, N_);
    gemm_relu_wmma<<<gemmBlocks, 128, 0, stream>>>(x, W1, bufA, N_, 1);
    node_scores_k<<<scoreBlocks, blk, 0, stream>>>(bufA, a_src1, a_dst1, asrc, adst, N_);
    edge_max_k<<<edgeBlocks, blk, 0, stream>>>(ei, asrc, adst, maxd, E_, N_);
    edge_exp_k<<<edgeBlocks, blk, 0, stream>>>(ei, asrc, adst, maxd, exbuf, denom, E_, N_);
    edge_agg_k<<<eaggBlocks, blk, 0, stream>>>(ei, bufA, exbuf, denom, bufB, E_, N_);
    bias_relu_k<<<2048, blk, 0, stream>>>(bufB, b1, (long)N_ * NF);   // bufB = h1

    // ---- layer 2 -----------------------------------------------------------
    gemm_relu_wmma<<<gemmBlocks, 128, 0, stream>>>(bufB, W2, bufA, N_, 0);  // bufA = h2_lin
    fill_f32<<<2048, blk, 0, stream>>>(bufB, 0.f, (long)N_ * NF);           // reuse as agg2
    fill_f32<<<512,  blk, 0, stream>>>(maxd, NEG_INF, N_);
    fill_f32<<<512,  blk, 0, stream>>>(denom, 0.f, N_);
    node_scores_k<<<scoreBlocks, blk, 0, stream>>>(bufA, a_src2, a_dst2, asrc, adst, N_);
    edge_max_k<<<edgeBlocks, blk, 0, stream>>>(ei, asrc, adst, maxd, E_, N_);
    edge_exp_k<<<edgeBlocks, blk, 0, stream>>>(ei, asrc, adst, maxd, exbuf, denom, E_, N_);
    edge_agg_k<<<eaggBlocks, blk, 0, stream>>>(ei, bufA, exbuf, denom, bufB, E_, N_);
    bias_relu_k<<<2048, blk, 0, stream>>>(bufB, b2, (long)N_ * NF);   // bufB = h2

    // ---- pooling + head ----------------------------------------------------
    fill_f32<<<64, blk, 0, stream>>>(maxp, NEG_INF, (long)G_ * NF);
    fill_f32<<<64, blk, 0, stream>>>(sump, 0.f, (long)G_ * NF);
    fill_f32<<<4,  blk, 0, stream>>>(cnt, 0.f, G_);
    count_k<<<(N_ + 255) / 256, blk, 0, stream>>>(batch, cnt, N_);
    pool_k<<<8192, blk, 0, stream>>>(bufB, batch, maxp, sump, N_);
    final_k<<<G_, blk, 0, stream>>>(maxp, sump, cnt, fc_w, fc_b, out);
}
